// SelfQueryHub_36069135352468
// MI455X (gfx1250) — matrix operand order repaired and checked by reading, not compiled, not run on hardware
//
#include <hip/hip_runtime.h>

// MI455X (gfx1250) spatial-reduction attention, flash-style, all GEMMs on WMMA.
// K/V tiles streamed into double-buffered LDS by the Tensor Data Mover:
// issue chunk i+1's DMA, s_wait_tensorcnt(2) so only the older pair must have
// landed, compute chunk i -> DMA overlaps WMMA + softmax.
//
// Roofline: ~12 GFLOP total, ~25 MB HBM traffic -> memory/launch bound.
// f16 activations + f32 WMMA accumulation; online softmax (no 655 MB attention
// matrix); ~10 MB workspace (fits easily in 192 MB L2).

typedef __attribute__((ext_vector_type(16))) _Float16 v16h;
typedef __attribute__((ext_vector_type(8)))  _Float16 v8h;
typedef __attribute__((ext_vector_type(8)))  float    v8f;
typedef __attribute__((ext_vector_type(4)))  unsigned v4u;
typedef __attribute__((ext_vector_type(4)))  int      v4i;
typedef __attribute__((ext_vector_type(8)))  int      v8i;

#define BATCH 2
#define CH    128
#define IMH   80
#define IMW   80
#define NPIX  (IMH*IMW)   // 6400
#define KVH   40
#define KVW   40
#define NKV   (KVH*KVW)   // 1600
#define HEADS 8
#define HD    16
#define QSCALE 0.25f      // HEAD_DIM^-0.5
#define BN_EPS 1e-5f

#if __has_builtin(__builtin_amdgcn_tensor_load_to_lds)
#define HAVE_TDM 1
#else
#define HAVE_TDM 0
#endif

#if HAVE_TDM
// Issue one TDM 2D tile load: `rows` x 16 f16 elements from a row-major
// (dim1 x 16) tensor into LDS at byte offset lds_off.
// D# bitfields per CDNA5 ISA.
static __device__ __forceinline__
void tdm_load_2d_f16(unsigned lds_off, const _Float16* g, int rows, int dim1)
{
    unsigned long long ga = (unsigned long long)(size_t)g;
    v4u g0 = (v4u){0u, 0u, 0u, 0u};
    g0[0] = 1u;                                         // count=1, user mode
    g0[1] = lds_off;                                    // lds_addr (bytes)
    g0[2] = (unsigned)ga;                               // global_addr[31:0]
    g0[3] = (unsigned)((ga >> 32) & 0x01FFFFFFull)      // global_addr[56:32]
          | (2u << 30);                                 // type = 2 ("image")
    v8i g1 = (v8i){0, 0, 0, 0, 0, 0, 0, 0};
    g1[0] = (int)(1u << 16);                            // data_size = 2 bytes
    g1[1] = (int)(16u << 16);                           // tensor_dim0 = 16
    g1[2] = (int)(((unsigned)dim1 & 0xFFFFu) << 16);    // tensor_dim1 lo
    g1[3] = (int)((((unsigned)dim1 >> 16) & 0xFFFFu)    // tensor_dim1 hi
          | (16u << 16));                               // tile_dim0 = 16
    g1[4] = (int)((unsigned)rows & 0xFFFFu);            // tile_dim1 = rows
    g1[5] = 16;                                         // tensor_dim0_stride
    v4i z4 = (v4i){0, 0, 0, 0};
#if defined(__clang_major__) && (__clang_major__ >= 23)
    v8i z8 = (v8i){0, 0, 0, 0, 0, 0, 0, 0};
    __builtin_amdgcn_tensor_load_to_lds(g0, g1, z4, z4, z8, 0);
#else
    __builtin_amdgcn_tensor_load_to_lds(g0, g1, z4, z4, 0);
#endif
}
#endif

// ---------------------------------------------------------------------------
// Stage 1: 2x2/s2 conv + folded BN + ReLU  ->  xkv (B, C, NKV) f32
// ---------------------------------------------------------------------------
__global__ __launch_bounds__(128)
void sr_bn_relu_kernel(const float* __restrict__ x,
                       const float* __restrict__ sr_w,
                       const float* __restrict__ sr_b,
                       const float* __restrict__ gamma,
                       const float* __restrict__ beta,
                       const float* __restrict__ mean,
                       const float* __restrict__ var,
                       float* __restrict__ xkv)
{
    const int p  = blockIdx.x % NKV;
    const int b  = blockIdx.x / NKV;
    const int rr = p / KVW, cc = p % KVW;
    const int o  = threadIdx.x;

    __shared__ float sx[CH * 4];
    {
        const float* xp = x + ((size_t)b*CH + threadIdx.x)*NPIX + (2*rr)*IMW + 2*cc;
        sx[threadIdx.x*4 + 0] = xp[0];
        sx[threadIdx.x*4 + 1] = xp[1];
        sx[threadIdx.x*4 + 2] = xp[IMW];
        sx[threadIdx.x*4 + 3] = xp[IMW + 1];
    }
    __syncthreads();

    const float* wr = sr_w + (size_t)o*CH*4;
    float acc = 0.f;
#pragma unroll 8
    for (int i = 0; i < CH*4; ++i) acc = fmaf(wr[i], sx[i], acc);

    const float inv = gamma[o] * rsqrtf(var[o] + BN_EPS);
    float v = (acc + sr_b[o]) * inv + (beta[o] - mean[o]*inv);
    xkv[((size_t)b*CH + o)*NKV + p] = fmaxf(v, 0.f);
}

// ---------------------------------------------------------------------------
// Stage 2: 1x1 conv as WMMA GEMM, M=128(out ch), K=128(in ch), N=pixels.
// B tile staged through LDS: coalesced float4 global loads, transposed store,
// contiguous v8h reads in exact B layout. Output per-head (B,HEADS,n,HD) f16.
// ---------------------------------------------------------------------------
__global__ __launch_bounds__(32)
void conv1x1_head_kernel(const float* __restrict__ xin,
                         const float* __restrict__ w,
                         const float* __restrict__ bias,
                         _Float16* __restrict__ outh,
                         int nn, float scale)
{
    const int lane = threadIdx.x;
    const int m0   = blockIdx.x * 16;
    const int n0   = blockIdx.y * 16;
    const int b    = blockIdx.z;
    const int grp  = lane >> 4;
    const int sub  = lane & 15;

    __shared__ alignas(16) _Float16 bt[16][40];  // [n][k], padded rows

    v8f c = {};
    for (int k0 = 0; k0 < CH; k0 += 32) {
        // Stage: lane owns K-row (k0+lane): 16 contiguous floats, cvt to f16,
        // transposed scatter into LDS.
        {
            const float4* xp = (const float4*)(xin + ((size_t)b*CH + k0 + lane)*nn + n0);
            float4 f0 = xp[0], f1 = xp[1], f2 = xp[2], f3 = xp[3];
            float fv[16] = { f0.x,f0.y,f0.z,f0.w, f1.x,f1.y,f1.z,f1.w,
                             f2.x,f2.y,f2.z,f2.w, f3.x,f3.y,f3.z,f3.w };
#pragma unroll
            for (int n = 0; n < 16; ++n) bt[n][lane] = (_Float16)fv[n];
        }
        __syncthreads();

        // A tile: weight rows, two contiguous 8-float runs {0..7},{16..23}+8*grp.
        v16h a;
        {
            const float4* wr = (const float4*)(w + (size_t)(m0 + sub)*CH + k0 + grp*8);
            float4 a0 = wr[0], a1 = wr[1], a2 = wr[4], a3 = wr[5];
            float av[16] = { a0.x,a0.y,a0.z,a0.w, a1.x,a1.y,a1.z,a1.w,
                             a2.x,a2.y,a2.z,a2.w, a3.x,a3.y,a3.z,a3.w };
#pragma unroll
            for (int j = 0; j < 16; ++j) a[j] = (_Float16)av[j];
        }
        // B tile from LDS: col = sub, K runs {0..7},{16..23}+8*grp.
        v8h blo = *(const v8h*)&bt[sub][grp*8];
        v8h bhi = *(const v8h*)&bt[sub][16 + grp*8];
        v16h bm = __builtin_shufflevector(blo, bhi,
                    0,1,2,3,4,5,6,7,8,9,10,11,12,13,14,15);

        c = __builtin_amdgcn_wmma_f32_16x16x32_f16(false, a, false, bm,
                                                   (short)0, c, false, false);
        __syncthreads();
    }

    const int head = m0 >> 4;
    v8h ov;
#pragma unroll
    for (int r = 0; r < 8; ++r)
        ov[r] = (_Float16)((c[r] + bias[m0 + grp*8 + r]) * scale);
    // Rows grp*8..grp*8+7 are contiguous halves -> one 16B store.
    *(v8h*)&outh[(((size_t)b*HEADS + head)*nn + n0 + sub)*HD + grp*8] = ov;
}

// ---------------------------------------------------------------------------
// Stage 3: flash attention. One wave per (b, h, 16-query tile).
// Double-buffered TDM streaming of K/V 32x16 tiles (s_wait_tensorcnt(2) keeps
// one pair in flight); S = Q x K^T and O += P x V on WMMA; online softmax with
// half-wave shfl reductions; P transposed D->A layout through LDS.
// ---------------------------------------------------------------------------
__global__ __launch_bounds__(32)
void attn_kernel(const _Float16* __restrict__ qh,
                 const _Float16* __restrict__ kh,
                 const _Float16* __restrict__ vh,
                 _Float16* __restrict__ ao)
{
    const int lane = threadIdx.x;
    const int n0   = blockIdx.x * 16;
    const int h    = blockIdx.y;
    const int b    = blockIdx.z;
    const int grp  = lane >> 4;
    const int sub  = lane & 15;
    const size_t bh = (size_t)b*HEADS + h;

#if HAVE_TDM
    __shared__ alignas(32) _Float16 kls[2][32][16];
    __shared__ alignas(32) _Float16 vls[2][32][16];
#endif
    __shared__ alignas(16) _Float16 psh[16][32];

    // Q tile in A layout; head dim zero-padded 16 -> K=32. QSCALE pre-folded.
    v16h aq;
    {
        const _Float16* qrow = qh + (bh*NPIX + n0 + sub)*HD;
        v8h q0 = *(const v8h*)qrow;
        v8h q1 = *(const v8h*)(qrow + 8);
#pragma unroll
        for (int j = 0; j < 16; ++j) {
            int k = j + grp*16;
            aq[j] = (k < HD) ? ((k < 8) ? q0[k] : q1[k - 8]) : (_Float16)0.f;
        }
    }

    float mrow[8], lrow[8];
#pragma unroll
    for (int r = 0; r < 8; ++r) { mrow[r] = -1e30f; lrow[r] = 0.f; }
    v8f acc = {};

#if HAVE_TDM
    // Prologue: DMA chunk 0 into buffer 0.
    tdm_load_2d_f16((unsigned)(size_t)&kls[0][0][0], kh + bh*NKV*HD, 32, NKV);
    tdm_load_2d_f16((unsigned)(size_t)&vls[0][0][0], vh + bh*NKV*HD, 32, NKV);
#endif

    int cur = 0;
    for (int kv0 = 0; kv0 < NKV; kv0 += 32, cur ^= 1) {
        v16h bk0 = {}, bk1 = {};
#if HAVE_TDM
        if (kv0 + 32 < NKV) {
            // DMA next chunk into the other buffer, then wait until only that
            // pair is still outstanding (older pair has landed).
            const _Float16* kn = kh + (bh*NKV + kv0 + 32)*HD;
            const _Float16* vn = vh + (bh*NKV + kv0 + 32)*HD;
            tdm_load_2d_f16((unsigned)(size_t)&kls[cur ^ 1][0][0], kn, 32, NKV);
            tdm_load_2d_f16((unsigned)(size_t)&vls[cur ^ 1][0][0], vn, 32, NKV);
#if __has_builtin(__builtin_amdgcn_s_wait_tensorcnt)
            __builtin_amdgcn_s_wait_tensorcnt(2);
#endif
        } else {
#if __has_builtin(__builtin_amdgcn_s_wait_tensorcnt)
            __builtin_amdgcn_s_wait_tensorcnt(0);
#endif
        }
        __syncthreads();
        if (grp == 0) {   // B layout: lanes 0-15 real K(=d), lanes 16-31 pad
            v8h a0 = *(const v8h*)&kls[cur][sub][0];
            v8h a1 = *(const v8h*)&kls[cur][sub][8];
            v8h b0 = *(const v8h*)&kls[cur][16 + sub][0];
            v8h b1 = *(const v8h*)&kls[cur][16 + sub][8];
            bk0 = __builtin_shufflevector(a0, a1, 0,1,2,3,4,5,6,7,8,9,10,11,12,13,14,15);
            bk1 = __builtin_shufflevector(b0, b1, 0,1,2,3,4,5,6,7,8,9,10,11,12,13,14,15);
        }
#else
        if (grp == 0) {
            const _Float16* kr0 = kh + (bh*NKV + kv0 + sub)*HD;
            v8h a0 = *(const v8h*)kr0,            a1 = *(const v8h*)(kr0 + 8);
            v8h b0 = *(const v8h*)(kr0 + 16*HD),  b1 = *(const v8h*)(kr0 + 16*HD + 8);
            bk0 = __builtin_shufflevector(a0, a1, 0,1,2,3,4,5,6,7,8,9,10,11,12,13,14,15);
            bk1 = __builtin_shufflevector(b0, b1, 0,1,2,3,4,5,6,7,8,9,10,11,12,13,14,15);
            if (kv0 + 32 < NKV)
                __builtin_prefetch(kh + (bh*NKV + kv0 + 32 + sub)*HD, 0, 0);
        }
#endif

        v8f s0 = {}, s1 = {};
        s0 = __builtin_amdgcn_wmma_f32_16x16x32_f16(false, aq, false, bk0,
                                                    (short)0, s0, false, false);
        s1 = __builtin_amdgcn_wmma_f32_16x16x32_f16(false, aq, false, bk1,
                                                    (short)0, s1, false, false);

        // Online softmax: row M = r + 8*grp in c-element r; 16 kv columns
        // across the 16-lane half-group -> xor-shuffle reductions.
#pragma unroll
        for (int r = 0; r < 8; ++r) {
            float mx = fmaxf(s0[r], s1[r]);
#pragma unroll
            for (int d = 1; d < 16; d <<= 1) mx = fmaxf(mx, __shfl_xor(mx, d, 32));
            float mnew  = fmaxf(mrow[r], mx);
            float alpha = __expf(mrow[r] - mnew);
            float p0 = __expf(s0[r] - mnew);
            float p1 = __expf(s1[r] - mnew);
            float ps = p0 + p1;
#pragma unroll
            for (int d = 1; d < 16; d <<= 1) ps += __shfl_xor(ps, d, 32);
            lrow[r] = lrow[r]*alpha + ps;
            mrow[r] = mnew;
            acc[r] *= alpha;
            int m = r + grp*8;
            psh[m][sub]      = (_Float16)p0;
            psh[m][16 + sub] = (_Float16)p1;
        }
        __syncthreads();

        // P tile in A layout: K(=kv) = grp*16 + j.
        v8h p0v = *(const v8h*)&psh[sub][grp*16];
        v8h p1v = *(const v8h*)&psh[sub][grp*16 + 8];
        v16h ap = __builtin_shufflevector(p0v, p1v,
                    0,1,2,3,4,5,6,7,8,9,10,11,12,13,14,15);

        // V chunk in B layout: col = d = sub, K(=kv) = grp*16 + j.
        v16h bv;
#if HAVE_TDM
#pragma unroll
        for (int j = 0; j < 16; ++j) bv[j] = vls[cur][grp*16 + j][sub];
#else
        {
            const _Float16* vp = vh + (bh*NKV + kv0 + grp*16)*HD + sub;
#pragma unroll
            for (int j = 0; j < 16; ++j) bv[j] = vp[j*HD];
        }
#endif
        acc = __builtin_amdgcn_wmma_f32_16x16x32_f16(false, ap, false, bv,
                                                     (short)0, acc, false, false);
        __syncthreads();   // protect psh (and LDS tiles) before next chunk
    }

    // Normalize; store with the reference's faithful reshape:
    // flat index i = (n*HEADS + h)*HD + d, later read back as (B, C, NPIX).
#pragma unroll
    for (int r = 0; r < 8; ++r) {
        int m = r + grp*8;
        float ov = acc[r] / lrow[r];
        ao[(size_t)b*CH*NPIX + ((size_t)(n0 + m)*HEADS + h)*HD + sub] = (_Float16)ov;
    }
}

// ---------------------------------------------------------------------------
// Stage 4: projection GEMM (f16 activations via LDS staging, f32 out).
// ---------------------------------------------------------------------------
__global__ __launch_bounds__(32)
void proj_kernel(const _Float16* __restrict__ ain,
                 const float* __restrict__ w,
                 const float* __restrict__ bias,
                 float* __restrict__ outp)
{
    const int lane = threadIdx.x;
    const int m0   = blockIdx.x * 16;
    const int n0   = blockIdx.y * 16;
    const int b    = blockIdx.z;
    const int grp  = lane >> 4;
    const int sub  = lane & 15;

    __shared__ alignas(16) _Float16 bt[16][40];

    v8f c = {};
    for (int k0 = 0; k0 < CH; k0 += 32) {
        {
            const v8h* xp = (const v8h*)(ain + ((size_t)b*CH + k0 + lane)*NPIX + n0);
            v8h h0 = xp[0], h1 = xp[1];
#pragma unroll
            for (int n = 0; n < 8; ++n)  bt[n][lane]     = h0[n];
#pragma unroll
            for (int n = 0; n < 8; ++n)  bt[n + 8][lane] = h1[n];
        }
        __syncthreads();

        v16h a;
        {
            const float4* wr = (const float4*)(w + (size_t)(m0 + sub)*CH + k0 + grp*8);
            float4 a0 = wr[0], a1 = wr[1], a2 = wr[4], a3 = wr[5];
            float av[16] = { a0.x,a0.y,a0.z,a0.w, a1.x,a1.y,a1.z,a1.w,
                             a2.x,a2.y,a2.z,a2.w, a3.x,a3.y,a3.z,a3.w };
#pragma unroll
            for (int j = 0; j < 16; ++j) a[j] = (_Float16)av[j];
        }
        v8h blo = *(const v8h*)&bt[sub][grp*8];
        v8h bhi = *(const v8h*)&bt[sub][16 + grp*8];
        v16h bm = __builtin_shufflevector(blo, bhi,
                    0,1,2,3,4,5,6,7,8,9,10,11,12,13,14,15);

        c = __builtin_amdgcn_wmma_f32_16x16x32_f16(false, a, false, bm,
                                                   (short)0, c, false, false);
        __syncthreads();
    }
#pragma unroll
    for (int r = 0; r < 8; ++r) {
        int m = r + grp*8;
        outp[((size_t)b*CH + m0 + m)*NPIX + n0 + sub] = c[r] + bias[m0 + m];
    }
}

// ---------------------------------------------------------------------------
extern "C" void kernel_launch(void* const* d_in, const int* in_sizes, int n_in,
                              void* d_out, int out_size, void* d_ws, size_t ws_size,
                              hipStream_t stream)
{
    const float* x      = (const float*)d_in[0];
    const float* q_w    = (const float*)d_in[1];
    const float* q_b    = (const float*)d_in[2];
    const float* k_w    = (const float*)d_in[3];
    const float* k_b    = (const float*)d_in[4];
    const float* v_w    = (const float*)d_in[5];
    const float* v_b    = (const float*)d_in[6];
    const float* sr_w   = (const float*)d_in[7];
    const float* sr_b   = (const float*)d_in[8];
    const float* gamma  = (const float*)d_in[9];
    const float* beta   = (const float*)d_in[10];
    const float* mean   = (const float*)d_in[11];
    const float* var    = (const float*)d_in[12];
    const float* proj_w = (const float*)d_in[13];
    const float* proj_b = (const float*)d_in[14];
    float* out = (float*)d_out;

    char* ws = (char*)d_ws;
    float*    xkv = (float*)ws;    ws += (size_t)BATCH*CH*NKV  * sizeof(float);
    _Float16* qh  = (_Float16*)ws; ws += (size_t)BATCH*CH*NPIX * sizeof(_Float16);
    _Float16* khb = (_Float16*)ws; ws += (size_t)BATCH*CH*NKV  * sizeof(_Float16);
    _Float16* vhb = (_Float16*)ws; ws += (size_t)BATCH*CH*NKV  * sizeof(_Float16);
    _Float16* ao  = (_Float16*)ws; ws += (size_t)BATCH*CH*NPIX * sizeof(_Float16);
    (void)ws_size; (void)in_sizes; (void)n_in; (void)out_size;

    sr_bn_relu_kernel<<<BATCH*NKV, CH, 0, stream>>>(x, sr_w, sr_b, gamma, beta,
                                                    mean, var, xkv);
    conv1x1_head_kernel<<<dim3(CH/16, NPIX/16, BATCH), 32, 0, stream>>>(
        x, q_w, q_b, qh, NPIX, QSCALE);
    conv1x1_head_kernel<<<dim3(CH/16, NKV/16, BATCH), 32, 0, stream>>>(
        xkv, k_w, k_b, khb, NKV, 1.0f);
    conv1x1_head_kernel<<<dim3(CH/16, NKV/16, BATCH), 32, 0, stream>>>(
        xkv, v_w, v_b, vhb, NKV, 1.0f);
    attn_kernel<<<dim3(NPIX/16, HEADS, BATCH), 32, 0, stream>>>(qh, khb, vhb, ao);
    proj_kernel<<<dim3(CH/16, NPIX/16, BATCH), 32, 0, stream>>>(ao, proj_w, proj_b, out);
}
